// MultiHeadAttention_48077863912223
// MI455X (gfx1250) — matrix operand order repaired and checked by reading, not compiled, not run on hardware
//
#include <hip/hip_runtime.h>

typedef __attribute__((ext_vector_type(16))) _Float16 v16h;
typedef __attribute__((ext_vector_type(8)))  float    v8f;
typedef __attribute__((ext_vector_type(4)))  unsigned int u32x4;
typedef __attribute__((ext_vector_type(8)))  int      i32x8;
typedef __attribute__((ext_vector_type(4)))  int      i32x4;

#define D_MODEL 1024
#define N_HEADS 16
#define DK      64
#define SEQ     2048
#define BATCH   4
#define LOG2E   1.4426950408889634f
#define GSTRIDE 40   // LDS row stride (halves) for 32-wide GEMM tiles (+8 pad)
#define ASTRIDE 72   // LDS row stride (halves) for 64-wide attention tiles (+8 pad)

__device__ __forceinline__ v8f wmma16(v16h a, v16h b, v8f c) {
  return __builtin_amdgcn_wmma_f32_16x16x32_f16(false, a, false, b, (short)0, c,
                                                false, false);
}

// A-matrix 16x32 f16 fragment (ISA 7.12.2): lanes 0-15 -> K{0..7,16..23},
// lanes 16-31 -> K{8..15,24..31}; rowp points at the 32-half K window of row M.
__device__ __forceinline__ v16h frag_a(const _Float16* rowp, int lane) {
  const int kb0 = (lane & 16) ? 8 : 0;
  union { uint4 u[2]; v16h v; } t;
  t.u[0] = *(const uint4*)(rowp + kb0);
  t.u[1] = *(const uint4*)(rowp + kb0 + 16);
  return t.v;
}
// B-matrix 32x16 f16 fragment, LDS holds B transposed (row = N, contiguous K):
// lanes 0-15 -> K{0..7,16..23}, lanes 16-31 -> K{8..15,24..31}.
__device__ __forceinline__ v16h frag_b(const _Float16* rowp, int lane) {
  const int kb = (lane & 16) ? 8 : 0;
  union { uint4 u[2]; v16h v; } t;
  t.u[0] = *(const uint4*)(rowp + kb);
  t.u[1] = *(const uint4*)(rowp + kb + 16);
  return t.v;
}

__device__ __forceinline__ void cvt_store16(_Float16* dst, float4 a, float4 b,
                                            float4 c, float4 d) {
  union { _Float16 h[16]; uint4 u[2]; } t;
  t.h[0]=(_Float16)a.x;  t.h[1]=(_Float16)a.y;  t.h[2]=(_Float16)a.z;  t.h[3]=(_Float16)a.w;
  t.h[4]=(_Float16)b.x;  t.h[5]=(_Float16)b.y;  t.h[6]=(_Float16)b.z;  t.h[7]=(_Float16)b.w;
  t.h[8]=(_Float16)c.x;  t.h[9]=(_Float16)c.y;  t.h[10]=(_Float16)c.z; t.h[11]=(_Float16)c.w;
  t.h[12]=(_Float16)d.x; t.h[13]=(_Float16)d.y; t.h[14]=(_Float16)d.z; t.h[15]=(_Float16)d.w;
  *(uint4*)dst       = t.u[0];
  *(uint4*)(dst + 8) = t.u[1];
}

// ---- Tensor Data Mover: DMA a 2D f16 tile (tile_d1 rows x tile_d0 elems)
// from global into LDS, padding each 128B row with 16B so LDS row stride is
// ASTRIDE halves (pad_interval code 4 = every 32 DWORDs, pad_amount code 3 =
// 4 DWORDs).  Descriptor per CDNA5 ISA section 8 (D# groups 0/1; 2D => groups
// 2/3 zero).  Issued per-wave (EXEC ignored), tracked on TENSORcnt.
__device__ __forceinline__ void tdm_load_2d_f16(
    unsigned lds_off, const void* gaddr, unsigned tile_d0, unsigned tile_d1,
    unsigned tensor_d0, unsigned tensor_d1, unsigned long long stride0) {
  const unsigned long long ga = (unsigned long long)(size_t)gaddr;
  u32x4 g0;
  g0[0] = 1u;                                              // count=1 (valid user D#)
  g0[1] = lds_off;                                         // lds_addr (bytes)
  g0[2] = (unsigned)(ga & 0xffffffffu);                    // global_addr[31:0]
  g0[3] = (unsigned)((ga >> 32) & 0x01ffffffu) | (2u << 30); // addr[56:32] | type=2
  i32x8 g1;
  g1[0] = (int)((1u << 16) | (1u << 20) | (4u << 22) | (3u << 25)); // 2B, pad cfg
  g1[1] = (int)((tensor_d0 & 0xffffu) << 16);              // tensor_dim0[15:0]
  g1[2] = (int)(((tensor_d0 >> 16) & 0xffffu) | ((tensor_d1 & 0xffffu) << 16));
  g1[3] = (int)(((tensor_d1 >> 16) & 0xffffu) | ((tile_d0 & 0xffffu) << 16));
  g1[4] = (int)(tile_d1 & 0xffffu);                        // tile_dim1 | tile_dim2=0
  g1[5] = (int)(stride0 & 0xffffffffull);                  // dim0_stride[31:0]
  g1[6] = (int)((stride0 >> 32) & 0xffffull);              // dim0_stride[47:32]
  g1[7] = 0;
  const i32x4 z4 = {0, 0, 0, 0};
  const i32x8 z8 = {0, 0, 0, 0, 0, 0, 0, 0};
  __builtin_amdgcn_tensor_load_to_lds(g0, g1, z4, z4, z8, 0);
}

// ---------------- Q/K/V projection: out = X @ W^T + bias ----------------
// X: [8192][1024] fp32 (rows m = s*BATCH+b).  W: [1024][1024] fp32.
// VMODE 0: out[b][h][s][dk] f16;  VMODE 1: out[b][h][dk][s] f16 (V transposed).
template <int VMODE>
__global__ __launch_bounds__(256) void proj_kernel(
    const float* __restrict__ X, const float* __restrict__ W,
    const float* __restrict__ bias, _Float16* __restrict__ out) {
  __shared__ _Float16 As[128 * GSTRIDE];
  __shared__ _Float16 Bs[128 * GSTRIDE];
  const int tid  = threadIdx.x;
  const int lane = tid & 31;
  const int wave = tid >> 5;
  const int m0 = blockIdx.y * 128;
  const int n0 = blockIdx.x * 128;
  const int wm = (wave >> 1) * 32;
  const int wn = (wave & 1) * 64;
  const int lrow = tid >> 1;
  const int lcol = (tid & 1) * 16;
  const int mrow = lane & 15;
  const int hi16 = (lane >> 4) & 1;
  const float* Ap = X + (size_t)(m0 + lrow) * D_MODEL + lcol;
  const float* Bp = W + (size_t)(n0 + lrow) * D_MODEL + lcol;
  _Float16* Asw = &As[lrow * GSTRIDE + lcol];
  _Float16* Bsw = &Bs[lrow * GSTRIDE + lcol];
  v8f acc[2][4] = {};
  for (int kk = 0; kk < D_MODEL; kk += 32) {
    float4 a0 = *(const float4*)(Ap + kk);
    float4 a1 = *(const float4*)(Ap + kk + 4);
    float4 a2 = *(const float4*)(Ap + kk + 8);
    float4 a3 = *(const float4*)(Ap + kk + 12);
    float4 b0 = *(const float4*)(Bp + kk);
    float4 b1 = *(const float4*)(Bp + kk + 4);
    float4 b2 = *(const float4*)(Bp + kk + 8);
    float4 b3 = *(const float4*)(Bp + kk + 12);
    __syncthreads();
    cvt_store16(Asw, a0, a1, a2, a3);
    cvt_store16(Bsw, b0, b1, b2, b3);
    __syncthreads();
    v16h af0 = frag_a(&As[(wm + mrow) * GSTRIDE], lane);
    v16h af1 = frag_a(&As[(wm + 16 + mrow) * GSTRIDE], lane);
    v16h bf0 = frag_b(&Bs[(wn + mrow) * GSTRIDE], lane);
    v16h bf1 = frag_b(&Bs[(wn + 16 + mrow) * GSTRIDE], lane);
    v16h bf2 = frag_b(&Bs[(wn + 32 + mrow) * GSTRIDE], lane);
    v16h bf3 = frag_b(&Bs[(wn + 48 + mrow) * GSTRIDE], lane);
    acc[0][0] = wmma16(af0, bf0, acc[0][0]);
    acc[0][1] = wmma16(af0, bf1, acc[0][1]);
    acc[0][2] = wmma16(af0, bf2, acc[0][2]);
    acc[0][3] = wmma16(af0, bf3, acc[0][3]);
    acc[1][0] = wmma16(af1, bf0, acc[1][0]);
    acc[1][1] = wmma16(af1, bf1, acc[1][1]);
    acc[1][2] = wmma16(af1, bf2, acc[1][2]);
    acc[1][3] = wmma16(af1, bf3, acc[1][3]);
  }
  #pragma unroll
  for (int mt = 0; mt < 2; ++mt)
    #pragma unroll
    for (int nt = 0; nt < 4; ++nt) {
      const int n = n0 + wn + nt * 16 + mrow;
      const float bv = bias[n];
      const int h = n >> 6, dk = n & 63;
      #pragma unroll
      for (int r = 0; r < 8; ++r) {
        const int m = m0 + wm + mt * 16 + r + 8 * hi16;
        const int b = m & 3, s = m >> 2;
        const float v = acc[mt][nt][r] + bv;
        size_t idx;
        if (VMODE) idx = ((size_t)(b * N_HEADS + h) * DK + dk) * SEQ + s;
        else       idx = ((size_t)(b * N_HEADS + h) * SEQ + s) * DK + dk;
        out[idx] = (_Float16)v;
      }
    }
}

// ---------------- flash attention: softmax(Q K^T / 8) V ----------------
__global__ __launch_bounds__(128) void attn_kernel(
    const _Float16* __restrict__ Qf, const _Float16* __restrict__ Kf,
    const _Float16* __restrict__ Vt, _Float16* __restrict__ Ctx) {
  __shared__ _Float16 Qs[128 * ASTRIDE];
  __shared__ _Float16 Ks[64 * ASTRIDE];
  __shared__ _Float16 Vs[64 * ASTRIDE];
  __shared__ _Float16 Ps[128 * ASTRIDE];
  const int tid = threadIdx.x, lane = tid & 31, wave = tid >> 5;
  const int bh = blockIdx.x;
  const int b = bh >> 4, h = bh & 15;
  const int q0 = blockIdx.y * 128;
  const int wm = wave * 32;
  const int mrow = lane & 15;
  const int hi16 = (lane >> 4) & 1;
  const _Float16* Qbase = Qf + ((size_t)bh * SEQ + q0) * DK;
  const _Float16* Kbase = Kf + (size_t)bh * SEQ * DK;
  const _Float16* Vbase = Vt + (size_t)bh * DK * SEQ;
  const unsigned qs_off = (unsigned)(size_t)(void*)Qs;
  const unsigned ks_off = (unsigned)(size_t)(void*)Ks;
  const unsigned vs_off = (unsigned)(size_t)(void*)Vs;

  // TDM: stage all 128 query rows once (64 x 128 f16 tile, row stride 64)
  if (wave == 0)
    tdm_load_2d_f16(qs_off, Qbase, DK, 128, DK, SEQ, DK);

  float mrun[2][8], lrun[2][8];
  v8f cacc[2][4] = {};
  #pragma unroll
  for (int mt = 0; mt < 2; ++mt)
    #pragma unroll
    for (int r = 0; r < 8; ++r) { mrun[mt][r] = -3.0e38f; lrun[mt][r] = 0.f; }

  for (int k0 = 0; k0 < SEQ; k0 += 64) {
    __syncthreads();  // all waves done reading previous K/V tiles
    if (wave == 0) {
      // K tile: 64 keys x 64 d (rows along s, contiguous d)
      tdm_load_2d_f16(ks_off, Kbase + (size_t)k0 * DK, DK, 64, DK, SEQ, DK);
      // V tile (pre-transposed): 64 dk-rows x 64 keys (row stride SEQ)
      tdm_load_2d_f16(vs_off, Vbase + k0, 64, DK, SEQ, DK, SEQ);
      __builtin_amdgcn_s_wait_tensorcnt(0);  // also covers the Q tile
    }
    __syncthreads();
    // ---- scores: S = Q K^T (two k=32 chunks over d) ----
    v16h qa[2][2];
    #pragma unroll
    for (int mt = 0; mt < 2; ++mt)
      #pragma unroll
      for (int c = 0; c < 2; ++c)
        qa[mt][c] = frag_a(&Qs[(wm + mt * 16 + mrow) * ASTRIDE + c * 32], lane);
    v8f sacc[2][4] = {};
    #pragma unroll
    for (int nt = 0; nt < 4; ++nt) {
      v16h kf0 = frag_b(&Ks[(nt * 16 + mrow) * ASTRIDE + 0], lane);
      v16h kf1 = frag_b(&Ks[(nt * 16 + mrow) * ASTRIDE + 32], lane);
      sacc[0][nt] = wmma16(qa[0][0], kf0, sacc[0][nt]);
      sacc[0][nt] = wmma16(qa[0][1], kf1, sacc[0][nt]);
      sacc[1][nt] = wmma16(qa[1][0], kf0, sacc[1][nt]);
      sacc[1][nt] = wmma16(qa[1][1], kf1, sacc[1][nt]);
    }
    // ---- online softmax (row = lane-group of 16; butterfly over 16 lanes) ----
    #pragma unroll
    for (int mt = 0; mt < 2; ++mt) {
      float mnew[8], alpha[8];
      #pragma unroll
      for (int nt = 0; nt < 4; ++nt)
        #pragma unroll
        for (int r = 0; r < 8; ++r) sacc[mt][nt][r] *= 0.125f;
      #pragma unroll
      for (int r = 0; r < 8; ++r) {
        float mx = fmaxf(fmaxf(sacc[mt][0][r], sacc[mt][1][r]),
                         fmaxf(sacc[mt][2][r], sacc[mt][3][r]));
        #pragma unroll
        for (int d = 1; d < 16; d <<= 1) mx = fmaxf(mx, __shfl_xor(mx, d, 32));
        const float mo = mrun[mt][r];
        const float mn = fmaxf(mo, mx);
        mnew[r] = mn;
        alpha[r] = exp2f((mo - mn) * LOG2E);
        mrun[mt][r] = mn;
      }
      #pragma unroll
      for (int r = 0; r < 8; ++r) {
        float p0 = exp2f((sacc[mt][0][r] - mnew[r]) * LOG2E);
        float p1 = exp2f((sacc[mt][1][r] - mnew[r]) * LOG2E);
        float p2 = exp2f((sacc[mt][2][r] - mnew[r]) * LOG2E);
        float p3 = exp2f((sacc[mt][3][r] - mnew[r]) * LOG2E);
        sacc[mt][0][r] = p0; sacc[mt][1][r] = p1;
        sacc[mt][2][r] = p2; sacc[mt][3][r] = p3;
        float rs = p0 + p1 + p2 + p3;
        #pragma unroll
        for (int d = 1; d < 16; d <<= 1) rs += __shfl_xor(rs, d, 32);
        lrun[mt][r] = lrun[mt][r] * alpha[r] + rs;
      }
      #pragma unroll
      for (int ct = 0; ct < 4; ++ct)
        #pragma unroll
        for (int r = 0; r < 8; ++r) cacc[mt][ct][r] *= alpha[r];
      // P -> wave-private LDS (f16) to re-fragment as A-matrix
      #pragma unroll
      for (int nt = 0; nt < 4; ++nt)
        #pragma unroll
        for (int r = 0; r < 8; ++r)
          Ps[(wm + mt * 16 + r + 8 * hi16) * ASTRIDE + nt * 16 + mrow] =
              (_Float16)sacc[mt][nt][r];
    }
    // ---- context += P V (two k=32 chunks over keys) ----
    #pragma unroll
    for (int c = 0; c < 2; ++c) {
      v16h pa0 = frag_a(&Ps[(wm + mrow) * ASTRIDE + c * 32], lane);
      v16h pa1 = frag_a(&Ps[(wm + 16 + mrow) * ASTRIDE + c * 32], lane);
      #pragma unroll
      for (int ct = 0; ct < 4; ++ct) {
        v16h vb = frag_b(&Vs[(ct * 16 + mrow) * ASTRIDE + c * 32], lane);
        cacc[0][ct] = wmma16(pa0, vb, cacc[0][ct]);
        cacc[1][ct] = wmma16(pa1, vb, cacc[1][ct]);
      }
    }
  }
  // ---- normalize and store context [b][s][h*64+dk] f16 ----
  #pragma unroll
  for (int mt = 0; mt < 2; ++mt) {
    float inv[8];
    #pragma unroll
    for (int r = 0; r < 8; ++r) inv[r] = 1.0f / lrun[mt][r];
    #pragma unroll
    for (int ct = 0; ct < 4; ++ct)
      #pragma unroll
      for (int r = 0; r < 8; ++r) {
        const int s = q0 + wm + mt * 16 + r + 8 * hi16;
        const int dcol = h * DK + ct * 16 + mrow;
        Ctx[((size_t)b * SEQ + s) * D_MODEL + dcol] =
            (_Float16)(cacc[mt][ct][r] * inv[r]);
      }
  }
}

// ---------------- output projection: out = Ctx @ Wo^T + bo ----------------
// Ctx rows m = b*SEQ+s (f16); out fp32 at [(s*BATCH+b)*D_MODEL + n].
__global__ __launch_bounds__(256) void outproj_kernel(
    const _Float16* __restrict__ A, const float* __restrict__ W,
    const float* __restrict__ bias, float* __restrict__ out) {
  __shared__ _Float16 As[128 * GSTRIDE];
  __shared__ _Float16 Bs[128 * GSTRIDE];
  const int tid  = threadIdx.x;
  const int lane = tid & 31;
  const int wave = tid >> 5;
  const int m0 = blockIdx.y * 128;
  const int n0 = blockIdx.x * 128;
  const int wm = (wave >> 1) * 32;
  const int wn = (wave & 1) * 64;
  const int lrow = tid >> 1;
  const int lcol = (tid & 1) * 16;
  const int mrow = lane & 15;
  const int hi16 = (lane >> 4) & 1;
  const _Float16* Ap = A + (size_t)(m0 + lrow) * D_MODEL + lcol;
  const float*    Bp = W + (size_t)(n0 + lrow) * D_MODEL + lcol;
  _Float16* Asw = &As[lrow * GSTRIDE + lcol];
  _Float16* Bsw = &Bs[lrow * GSTRIDE + lcol];
  v8f acc[2][4] = {};
  for (int kk = 0; kk < D_MODEL; kk += 32) {
    uint4 a0 = *(const uint4*)(Ap + kk);
    uint4 a1 = *(const uint4*)(Ap + kk + 8);
    float4 b0 = *(const float4*)(Bp + kk);
    float4 b1 = *(const float4*)(Bp + kk + 4);
    float4 b2 = *(const float4*)(Bp + kk + 8);
    float4 b3 = *(const float4*)(Bp + kk + 12);
    __syncthreads();
    *(uint4*)Asw = a0;
    *(uint4*)(Asw + 8) = a1;
    cvt_store16(Bsw, b0, b1, b2, b3);
    __syncthreads();
    v16h af0 = frag_a(&As[(wm + mrow) * GSTRIDE], lane);
    v16h af1 = frag_a(&As[(wm + 16 + mrow) * GSTRIDE], lane);
    v16h bf0 = frag_b(&Bs[(wn + mrow) * GSTRIDE], lane);
    v16h bf1 = frag_b(&Bs[(wn + 16 + mrow) * GSTRIDE], lane);
    v16h bf2 = frag_b(&Bs[(wn + 32 + mrow) * GSTRIDE], lane);
    v16h bf3 = frag_b(&Bs[(wn + 48 + mrow) * GSTRIDE], lane);
    acc[0][0] = wmma16(af0, bf0, acc[0][0]);
    acc[0][1] = wmma16(af0, bf1, acc[0][1]);
    acc[0][2] = wmma16(af0, bf2, acc[0][2]);
    acc[0][3] = wmma16(af0, bf3, acc[0][3]);
    acc[1][0] = wmma16(af1, bf0, acc[1][0]);
    acc[1][1] = wmma16(af1, bf1, acc[1][1]);
    acc[1][2] = wmma16(af1, bf2, acc[1][2]);
    acc[1][3] = wmma16(af1, bf3, acc[1][3]);
  }
  #pragma unroll
  for (int mt = 0; mt < 2; ++mt)
    #pragma unroll
    for (int nt = 0; nt < 4; ++nt) {
      const int n = n0 + wn + nt * 16 + mrow;
      const float bv = bias[n];
      #pragma unroll
      for (int r = 0; r < 8; ++r) {
        const int m = m0 + wm + mt * 16 + r + 8 * hi16;
        const int bb = m >> 11, s = m & 2047;
        out[((size_t)s * BATCH + bb) * D_MODEL + n] = acc[mt][nt][r] + bv;
      }
    }
}

extern "C" void kernel_launch(void* const* d_in, const int* in_sizes, int n_in,
                              void* d_out, int out_size, void* d_ws, size_t ws_size,
                              hipStream_t stream) {
  const float* q  = (const float*)d_in[0];
  const float* k  = (const float*)d_in[1];
  const float* v  = (const float*)d_in[2];
  const float* Wq = (const float*)d_in[3];
  const float* bq = (const float*)d_in[4];
  const float* Wk = (const float*)d_in[5];
  const float* bk = (const float*)d_in[6];
  const float* Wv = (const float*)d_in[7];
  const float* bv = (const float*)d_in[8];
  const float* Wo = (const float*)d_in[9];
  const float* bo = (const float*)d_in[10];
  float* out = (float*)d_out;

  const size_t HEADS_ELEMS = (size_t)BATCH * N_HEADS * SEQ * DK;  // 8M halves
  _Float16* qf  = (_Float16*)d_ws;
  _Float16* kf  = qf + HEADS_ELEMS;
  _Float16* vt  = kf + HEADS_ELEMS;
  _Float16* ctx = vt + HEADS_ELEMS;

  dim3 gridG(D_MODEL / 128, (SEQ * BATCH) / 128);  // (8, 64)
  proj_kernel<0><<<gridG, 256, 0, stream>>>(q, Wq, bq, qf);
  proj_kernel<0><<<gridG, 256, 0, stream>>>(k, Wk, bk, kf);
  proj_kernel<1><<<gridG, 256, 0, stream>>>(v, Wv, bv, vt);  // V stored transposed

  dim3 gridA(BATCH * N_HEADS, SEQ / 128);  // (64, 16)
  attn_kernel<<<gridA, 128, 0, stream>>>(qf, kf, vt, ctx);

  outproj_kernel<<<gridG, 256, 0, stream>>>(ctx, Wo, bo, out);
}
